// RNN_BN_33603824123985
// MI455X (gfx1250) — matrix operand order repaired and checked by reading, not compiled, not run on hardware
//
#include <hip/hip_runtime.h>

// Problem constants (match reference)
#define BB   131072
#define TT   30
#define DD   10
#define HH   20
#define EPSI 1e-5f
#define NBLK 1024      // layer-pass blocks: BB / (8 waves * 16 rows)
#define XCH  16        // chunks for x-stats partial reduction

typedef __attribute__((ext_vector_type(16))) _Float16 v16h;
typedef __attribute__((ext_vector_type(8)))  float    v8f;

// Load 20 consecutive floats (16B-aligned) as 5x b128 into registers.
__device__ __forceinline__ void load20(const float* __restrict__ p, float* r) {
  const float4* q = (const float4*)p;
  float4 a0 = q[0], a1 = q[1], a2 = q[2], a3 = q[3], a4 = q[4];
  r[0]=a0.x; r[1]=a0.y; r[2]=a0.z; r[3]=a0.w;
  r[4]=a1.x; r[5]=a1.y; r[6]=a1.z; r[7]=a1.w;
  r[8]=a2.x; r[9]=a2.y; r[10]=a2.z; r[11]=a2.w;
  r[12]=a3.x; r[13]=a3.y; r[14]=a3.z; r[15]=a3.w;
  r[16]=a4.x; r[17]=a4.y; r[18]=a4.z; r[19]=a4.w;
}

// Load 10 consecutive floats (8B-aligned) as 5x b64.
__device__ __forceinline__ void load10(const float* __restrict__ p, float* r) {
  const float2* q = (const float2*)p;
  float2 a0 = q[0], a1 = q[1], a2 = q[2], a3 = q[3], a4 = q[4];
  r[0]=a0.x; r[1]=a0.y; r[2]=a1.x; r[3]=a1.y; r[4]=a2.x;
  r[5]=a2.y; r[6]=a3.x; r[7]=a3.y; r[8]=a4.x; r[9]=a4.y;
}

// ---------------------------------------------------------------------------
// Pass 0: per-(t,d) column stats of x (input-only -> once for all 30 steps).
// ---------------------------------------------------------------------------
__global__ __launch_bounds__(256)
void xstats_part_kernel(const float* __restrict__ x, float* __restrict__ xpart) {
  const int t   = blockIdx.x / XCH;
  const int ch  = blockIdx.x % XCH;
  const int tid = threadIdx.x;
  const int rows = BB / XCH;
  float s[DD], q[DD];
#pragma unroll
  for (int d = 0; d < DD; ++d) { s[d] = 0.f; q[d] = 0.f; }
  for (int b = ch * rows + tid; b < (ch + 1) * rows; b += 256) {
    float rv[DD];
    load10(x + (size_t)b * (TT * DD) + t * DD, rv);
#pragma unroll
    for (int d = 0; d < DD; ++d) { s[d] += rv[d]; q[d] += rv[d] * rv[d]; }
  }
  __shared__ float sm[512];
  for (int d = 0; d < DD; ++d) {
    sm[tid] = s[d]; sm[256 + tid] = q[d];
    __syncthreads();
    for (int off = 128; off > 0; off >>= 1) {
      if (tid < off) { sm[tid] += sm[tid + off]; sm[256 + tid] += sm[256 + tid + off]; }
      __syncthreads();
    }
    if (tid == 0) {
      int c = (t * XCH + ch) * DD + d;
      xpart[2 * c] = sm[0]; xpart[2 * c + 1] = sm[256];
    }
    __syncthreads();
  }
}

__global__ void xstats_fin_kernel(const float* __restrict__ xpart,
                                  float* __restrict__ xstats) {
  const int c = threadIdx.x;           // c = t*DD + d
  if (c >= TT * DD) return;
  const int t = c / DD, d = c % DD;
  float S = 0.f, Q = 0.f;
  for (int ch = 0; ch < XCH; ++ch) {
    int cc = (t * XCH + ch) * DD + d;
    S += xpart[2 * cc]; Q += xpart[2 * cc + 1];
  }
  const float mean = S * (1.f / (float)BB);
  const float var  = fmaxf(Q * (1.f / (float)BB) - mean * mean, 0.f);
  xstats[2 * c] = mean; xstats[2 * c + 1] = var;
}

// ---------------------------------------------------------------------------
// BN finalize (layers 1,2): fixed-order reduce of block partials -> affine.
// ---------------------------------------------------------------------------
__global__ void bn_fin_kernel(const float* __restrict__ partials, int ncols,
                              const float* __restrict__ g,
                              const float* __restrict__ be,
                              float* __restrict__ aff) {
  const int c = threadIdx.x;
  if (c >= ncols) return;
  float S = 0.f, Q = 0.f;
  for (int bkt = 0; bkt < NBLK; ++bkt) {
    S += partials[bkt * 40 + 2 * c];
    Q += partials[bkt * 40 + 2 * c + 1];
  }
  const float mean = S * (1.f / (float)BB);
  const float var  = fmaxf(Q * (1.f / (float)BB) - mean * mean, 0.f);
  const float a = g[c] * rsqrtf(var + EPSI);
  aff[2 * c] = a; aff[2 * c + 1] = be[c] - mean * a;
}

// ---------------------------------------------------------------------------
// BN0 finalize + weight folding. No ReLU between BN0 and the layer-1 matmul:
//   BN0(h)@W1^T + b1 == h@(a.*W1)^T + (b1 + W1@c)
// so emit scaled W1' and folded bias; layer-1 A fragments become raw loads.
// Prev-column stats come from the previous step's output partials
// (usepart=0 at t=0: prev is identically zero -> mean=var=0).
// ---------------------------------------------------------------------------
__global__ void bn0_fold_kernel(const float* __restrict__ partials, int usepart,
                                const float* __restrict__ xst,
                                const float* __restrict__ g,
                                const float* __restrict__ be,
                                const float* __restrict__ W1t,
                                const float* __restrict__ b1t,
                                float* __restrict__ Ws,
                                float* __restrict__ bf) {
  __shared__ float aS[2 * DD], cS[2 * DD];
  const int c = threadIdx.x;
  if (c < 2 * DD) {
    float mean, var;
    if (c < DD) { mean = xst[2 * c]; var = xst[2 * c + 1]; }
    else if (usepart) {
      const int col = c - DD;
      float S = 0.f, Q = 0.f;
      for (int bkt = 0; bkt < NBLK; ++bkt) {
        S += partials[bkt * 40 + 2 * col];
        Q += partials[bkt * 40 + 2 * col + 1];
      }
      mean = S * (1.f / (float)BB);
      var  = fmaxf(Q * (1.f / (float)BB) - mean * mean, 0.f);
    } else { mean = 0.f; var = 0.f; }
    const float a = g[c] * rsqrtf(var + EPSI);
    aS[c] = a; cS[c] = be[c] - mean * a;
  }
  __syncthreads();
  if (c < HH) {
    float acc = b1t[c];
#pragma unroll
    for (int k = 0; k < 2 * DD; ++k) {
      const float w = W1t[c * (2 * DD) + k];
      Ws[c * (2 * DD) + k] = aS[k] * w;
      acc += cS[k] * w;
    }
    bf[c] = acc;
  }
}

// ---------------------------------------------------------------------------
// Layer pass: per wave, one 16-row batch tile.
//   D[16x16] = A[16x32] * B[32x16] (+bias) via v_wmma_f32_16x16x32_f16
// MODE 0: A = raw concat(x_t, prev) (BN0 folded into W'),  out u1 (N=20)
// MODE 1: A = relu(aff1(u1)),                              out u2 (N=20)
// MODE 2: A = relu(aff2(u2)), out = d_out[:,t,:] + prevbuf        (N=10)
// A/B fragments built from dense b128/b64 row loads + branchless lane-half
// selects (contiguous K = 16*hi + slot order). u1/u2 written via LDS-staged coalesced b128.
// Column (sum,sumsq) partials harvested from D fragments, deterministic LDS
// tree order -> bitwise-identical replays.
// ---------------------------------------------------------------------------
template <int MODE, int NDIM>
__global__ __launch_bounds__(256)
void layer_kernel(int t, int haveprev,
                  const float* __restrict__ x, const float* __restrict__ prevbuf,
                  const float* __restrict__ uin,
                  const float* __restrict__ W, const float* __restrict__ bias,
                  const float* __restrict__ aff,
                  float* __restrict__ uout, float* __restrict__ prevout,
                  float* __restrict__ partials) {
  constexpr int KDIM = 20;                 // 2D or H, both 20; padded to K=32
  const int tid = threadIdx.x;
  const int l   = tid & 31;
  const int wv  = tid >> 5;
  const int m   = l & 15;
  const int hi  = (l >= 16) ? 1 : 0;
  const int rowbase = blockIdx.x * 128 + wv * 16;

  // ---- A source row (20 floats) + optional BN affine table, all in regs.
  const int arow = rowbase + m;
  float rowv[KDIM];
  float aftab[2 * KDIM];
  if (MODE == 0) {
    load10(x + (size_t)arow * (TT * DD) + t * DD, rowv);
    if (haveprev) load10(prevbuf + (size_t)arow * DD, rowv + DD);
    else {
#pragma unroll
      for (int j = 0; j < DD; ++j) rowv[DD + j] = 0.f;
    }
  } else {
    load20(uin + (size_t)arow * HH, rowv);
    load20(aff, aftab);
    load20(aff + KDIM, aftab + KDIM);
  }

  // ---- A fragment (16-bit A 16x32): lane-half hi selects the K sub-set.
  //  hi=0: slots 0..15 -> K=i
  //  hi=1: slots 0..3 -> K=i+16, slots 4..15 -> K>=20 (invalid)
  v16h a;
#pragma unroll
  for (int i = 0; i < 16; ++i) {
    float v0 = 0.f, v1 = 0.f;
    if (i < 16) {                       // variant hi=0, K = i
      const int k = i;
      float r = rowv[k];
      if (MODE != 0) r = fmaxf(r * aftab[2 * k] + aftab[2 * k + 1], 0.f);
      v0 = r;
    }
    if (i < 4) {                        // variant hi=1, K = i+16
      const int k = i + 16;
      float r = rowv[k];
      if (MODE != 0) r = fmaxf(r * aftab[2 * k] + aftab[2 * k + 1], 0.f);
      v1 = r;
    }
    a[i] = (_Float16)(hi ? v1 : v0);
  }

  // ---- B fragments: B[k][n] = W[n][k]; lane n = m (tile0) / m+16 (tile1).
  //  slot i: K = hi*16 + i  -> hi=0: W row elems [0..15]; hi=1: [16..19].
  const int  n0v = (NDIM >= 16) || (m < NDIM);
  const int  n0c = (NDIM >= 16) ? m : (m < NDIM ? m : NDIM - 1);
  float w0r[KDIM];
  load20(W + n0c * KDIM, w0r);
  v16h b0;
#pragma unroll
  for (int i = 0; i < 16; ++i) {
    const float u0 = w0r[i];
    const float u1 = (i < 4) ? w0r[16 + i] : 0.f;
    b0[i] = (_Float16)(n0v ? (hi ? u1 : u0) : 0.f);
  }

  v8f c0 = {};
  c0 = __builtin_amdgcn_wmma_f32_16x16x32_f16(false, a, false, b0, (short)0, c0,
                                              false, false);
  v8f c1 = {};
  if (NDIM > 16) {
    const int n1v = (m + 16) < NDIM;
    const int n1c = n1v ? (m + 16) : (NDIM - 1);
    float w1r[KDIM];
    load20(W + n1c * KDIM, w1r);
    v16h b1f;
#pragma unroll
    for (int i = 0; i < 16; ++i) {
      const float u0 = w1r[i];
      const float u1 = (i < 4) ? w1r[16 + i] : 0.f;
      b1f[i] = (_Float16)(n1v ? (hi ? u1 : u0) : 0.f);
    }
    c1 = __builtin_amdgcn_wmma_f32_16x16x32_f16(false, a, false, b1f, (short)0, c1,
                                                false, false);
  }

  // ---- Epilogue. D layout: lane l, reg r -> row = r + 8*hi, col = m (+16).
  constexpr int STG = (MODE == 2) ? 4 : 128 * 20;
  __shared__ float stg[STG];
  __shared__ float sred[8][32][4];
  const int lrow = wv * 16 + hi * 8;
  const int n0 = m, n1 = m + 16;
  const float bi0 = ((NDIM >= 16) || (n0 < NDIM)) ? bias[n0 < NDIM ? n0 : 0] : 0.f;
  float s0 = 0.f, q0 = 0.f, s1 = 0.f, q1 = 0.f;
#pragma unroll
  for (int r = 0; r < 8; ++r) {
    if (MODE == 2) {
      if (n0 < NDIM) {
        const size_t rr = (size_t)(rowbase + hi * 8 + r);
        const float dv = c0[r] + bi0;
        uout[rr * (TT * DD) + t * DD + n0] = dv;   // final output [B,T,D]
        prevout[rr * DD + n0] = dv;                // recurrent feed
        s0 += dv; q0 += dv * dv;
      }
    } else {
      const float dv = c0[r] + bi0;                // n0 = m < 16 < NDIM: valid
      stg[(lrow + r) * 20 + n0] = dv;
      s0 += dv; q0 += dv * dv;
    }
  }
  if (NDIM > 16) {
    if (m < NDIM - 16) {
      const float bi1 = bias[n1];
#pragma unroll
      for (int r = 0; r < 8; ++r) {
        const float dv = c1[r] + bi1;
        stg[(lrow + r) * 20 + n1] = dv;
        s1 += dv; q1 += dv * dv;
      }
    }
  }

  sred[wv][l][0] = s0; sred[wv][l][1] = q0; sred[wv][l][2] = s1; sred[wv][l][3] = q1;
  __syncthreads();

  if (MODE != 2) {
    // Coalesced b128 writeback of the whole 128x20 tile.
    float4* gp = (float4*)(uout + (size_t)blockIdx.x * (128 * 20));
    const float4* sp = (const float4*)stg;
#pragma unroll
    for (int j = 0; j < 640 / 256 + 1; ++j) {
      const int idx = tid + j * 256;
      if (idx < 640) gp[idx] = sp[idx];
    }
  }

  if (tid < NDIM) {
    const int c  = tid;
    const int fb = (c < 16) ? 0 : 2;
    const int lc = c & 15;
    float S = 0.f, Q = 0.f;
    for (int w = 0; w < 8; ++w) {
      S += sred[w][lc][fb]     + sred[w][lc + 16][fb];
      Q += sred[w][lc][fb + 1] + sred[w][lc + 16][fb + 1];
    }
    partials[blockIdx.x * 40 + 2 * c]     = S;
    partials[blockIdx.x * 40 + 2 * c + 1] = Q;
  }
}

// ---------------------------------------------------------------------------
extern "C" void kernel_launch(void* const* d_in, const int* in_sizes, int n_in,
                              void* d_out, int out_size, void* d_ws, size_t ws_size,
                              hipStream_t stream) {
  const float* x   = (const float*)d_in[0];
  const float* g0  = (const float*)d_in[1];
  const float* be0 = (const float*)d_in[2];
  const float* W1  = (const float*)d_in[3];
  const float* b1  = (const float*)d_in[4];
  const float* g1  = (const float*)d_in[5];
  const float* be1 = (const float*)d_in[6];
  const float* W2  = (const float*)d_in[7];
  const float* b2  = (const float*)d_in[8];
  const float* g2  = (const float*)d_in[9];
  const float* be2 = (const float*)d_in[10];
  const float* W3  = (const float*)d_in[11];
  const float* b3  = (const float*)d_in[12];
  float* out = (float*)d_out;
  float* ws  = (float*)d_ws;

  // Workspace layout (floats); total ~26.4 MB, L2-resident on MI455X.
  float* xpart    = ws;                                  // TT*XCH*DD*2 = 9600
  float* xstats   = xpart + TT * XCH * DD * 2;           // TT*DD*2     = 600
  float* a1c1     = xstats + TT * DD * 2;                // 40
  float* a2c2     = a1c1 + 40;                           // 40
  float* W1s      = a2c2 + 40;                           // HH*2*DD = 400
  float* b1f      = W1s + HH * 2 * DD;                   // HH = 20
  float* partials = b1f + HH;                            // NBLK*40
  float* u1       = partials + NBLK * 40;                // BB*HH
  float* u2       = u1 + (size_t)BB * HH;                // BB*HH
  float* prevbuf  = u2 + (size_t)BB * HH;                // BB*DD

  // One-time x column statistics (all 30 steps).
  xstats_part_kernel<<<TT * XCH, 256, 0, stream>>>(x, xpart);
  xstats_fin_kernel<<<1, 320, 0, stream>>>(xpart, xstats);

  for (int t = 0; t < TT; ++t) {
    if (t == 0)
      bn0_fold_kernel<<<1, 32, 0, stream>>>(partials, 0, xstats, g0, be0,
                                            W1, b1, W1s, b1f);

    layer_kernel<0, HH><<<NBLK, 256, 0, stream>>>(
        t, (t > 0) ? 1 : 0, x, prevbuf, nullptr,
        W1s, b1f, nullptr, u1, nullptr, partials);
    bn_fin_kernel<<<1, 32, 0, stream>>>(partials, HH, g1 + t * HH, be1 + t * HH, a1c1);

    layer_kernel<1, HH><<<NBLK, 256, 0, stream>>>(
        t, 0, nullptr, nullptr, u1,
        W2 + (size_t)t * HH * HH, b2 + t * HH, a1c1, u2, nullptr, partials);
    bn_fin_kernel<<<1, 32, 0, stream>>>(partials, HH, g2 + t * HH, be2 + t * HH, a2c2);

    layer_kernel<2, DD><<<NBLK, 256, 0, stream>>>(
        t, 0, nullptr, nullptr, u2,
        W3 + (size_t)t * DD * HH, b3 + t * DD, a2c2, out, prevbuf, partials);

    if (t + 1 < TT)
      bn0_fold_kernel<<<1, 32, 0, stream>>>(
          partials, 1, xstats + (t + 1) * DD * 2,
          g0 + (size_t)(t + 1) * 2 * DD, be0 + (size_t)(t + 1) * 2 * DD,
          W1 + (size_t)(t + 1) * HH * 2 * DD, b1 + (size_t)(t + 1) * HH,
          W1s, b1f);
  }
}
